// MultiheadAttention_FFN_25305947308180
// MI455X (gfx1250) — compile-verified, matrix operand order checked by reading, not run on hardware
//
#include <hip/hip_runtime.h>
#include <hip/hip_bf16.h>
#include <math.h>

// ---------------------------------------------------------------------------
// MI455X / gfx1250 transformer block: QKV gemm -> flash attention -> SiLU FFN
// All matmuls run on v_wmma_f32_16x16x32_bf16 (bf16 inputs, f32 accumulate).
// K-tile staging in attention uses GLOBAL_LOAD_ASYNC_TO_LDS_B128 (ASYNCcnt).
// ---------------------------------------------------------------------------

typedef __attribute__((ext_vector_type(16))) __bf16 v16bf;
typedef __attribute__((ext_vector_type(8)))  __bf16 v8bf;
typedef __attribute__((ext_vector_type(8)))  float  v8f;
typedef __attribute__((ext_vector_type(4)))  float  v4f;

#define TDIM   1024
#define THEADS 16
#define TBS    2
#define TSEQ   2048
#define THD    64
#define TROWS  (TBS * TSEQ)   // 4096

// Pack 8 consecutive f32 -> v8bf
__device__ __forceinline__ v8bf cvt8(const float* p)
{
    v4f f0 = *(const v4f*)p;
    v4f f1 = *(const v4f*)(p + 4);
    v8bf r;
#pragma unroll
    for (int j = 0; j < 4; ++j) { r[j] = (__bf16)f0[j]; r[4 + j] = (__bf16)f1[j]; }
    return r;
}

// Load a 16x32 bf16 WMMA fragment (A-layout; identical addressing for B when
// the source is stored [col][k]). Per ISA 16-bit A layout:
//   lane<16 : elems 0..7 = K 0..7,  elems 8..15 = K 16..23   (row = lane)
//   lane>=16: elems 0..7 = K 8..15, elems 8..15 = K 24..31   (row = lane-16)
// Row strides used below (40, 72 elems) are multiples of 8 -> 16B aligned.
__device__ __forceinline__ v16bf load_frag(const __bf16* base, int stride,
                                           int k0, int lane)
{
    const int half = (lane >> 4) & 1;
    const int r    = lane & 15;
    const __bf16* p = base + r * stride + k0 + half * 8;
    v8bf lo = *(const v8bf*)p;
    v8bf hi = *(const v8bf*)(p + 16);
    return __builtin_shufflevector(lo, hi, 0, 1, 2, 3, 4, 5, 6, 7,
                                           8, 9, 10, 11, 12, 13, 14, 15);
}

// ---------------------------------------------------------------------------
// Generic GEMM: C[M,N] = act(A[M,K] @ B[K,N] + bias)
// f32 inputs converted to bf16 while staging to LDS. 64x64 block tile,
// 8 waves (4 along M x 2 along N), each wave -> 16x32 via two WMMAs / K-step.
// ---------------------------------------------------------------------------
template <int ACT, int OUT_BF16, int HAS_BIAS>
__global__ __launch_bounds__(256)
void gemm_wmma_kernel(const float* __restrict__ A, const float* __restrict__ B,
                      const float* __restrict__ bias, void* __restrict__ Cout,
                      int M, int N, int K)
{
    __shared__ __bf16 As[64][40];   // [m][k], +8 pad
    __shared__ __bf16 Bs[64][40];   // [n][k] (transposed), +8 pad

    const int tid  = threadIdx.x;
    const int wave = tid >> 5;
    const int lane = tid & 31;
    const int wm   = wave & 3;      // M subtile 0..3
    const int wn   = wave >> 2;     // N half    0..1
    const int m0   = blockIdx.x * 64;
    const int n0   = blockIdx.y * 64;

    // staging coordinates (one pass per tile, 8 contiguous elems per thread)
    const int ar  = tid >> 2;            // A: row 0..63
    const int ac  = (tid & 3) * 8;       // A: k-chunk 0..24
    const int bk  = tid >> 3;            // B: k-row 0..31
    const int bc  = (tid & 7) * 8;       // B: n-chunk 0..56

    v8f acc[2] = {};

    for (int k0 = 0; k0 < K; k0 += 32) {
        // stage A tile 64x32: vector load + pack + single b128 LDS store
        *(v8bf*)&As[ar][ac] = cvt8(&A[(size_t)(m0 + ar) * K + k0 + ac]);

        // stage B tile 32x64 transposed into [n][k] (scalar stores: transpose)
        {
            v8bf pb = cvt8(&B[(size_t)(k0 + bk) * N + n0 + bc]);
#pragma unroll
            for (int j = 0; j < 8; ++j) Bs[bc + j][bk] = pb[j];
        }
        __syncthreads();

        v16bf a  = load_frag(&As[wm * 16][0], 40, 0, lane);
        v16bf b0 = load_frag(&Bs[wn * 32][0], 40, 0, lane);
        v16bf b1 = load_frag(&Bs[wn * 32 + 16][0], 40, 0, lane);
        acc[0] = __builtin_amdgcn_wmma_f32_16x16x32_bf16(false, a, false, b0,
                                                         (short)0, acc[0], false, false);
        acc[1] = __builtin_amdgcn_wmma_f32_16x16x32_bf16(false, a, false, b1,
                                                         (short)0, acc[1], false, false);
        __syncthreads();
    }

    // epilogue: D layout -> lane holds col (lane&15); slot i -> row i (+8 if lane>=16)
    const int half = lane >> 4, nl = lane & 15;
#pragma unroll
    for (int t = 0; t < 2; ++t) {
        const int n = n0 + wn * 32 + t * 16 + nl;
        const float bv = HAS_BIAS ? bias[n] : 0.0f;
#pragma unroll
        for (int i = 0; i < 8; ++i) {
            const int m = m0 + wm * 16 + i + 8 * half;
            float x = acc[t][i] + bv;
            if (ACT == 1)  // SiLU: x * sigmoid(x), v_rcp_f32 instead of IEEE div
                x = x * __builtin_amdgcn_rcpf(1.0f + __expf(-x));
            if (OUT_BF16) ((__bf16*)Cout)[(size_t)m * N + n] = (__bf16)x;
            else          ((float*)Cout)[(size_t)m * N + n]  = x;
        }
    }
}

// ---------------------------------------------------------------------------
// Flash attention: one block per (batch*head, 128 q rows).
// Each wave owns 16 q rows; streams 32-key tiles through LDS; online softmax.
// qkv is bf16 [4096, 3072]: Q at +0, K at +1024, V at +2048 (per row, per head).
// K tiles are staged with the CDNA5 async global->LDS DMA path (ASYNCcnt).
// ---------------------------------------------------------------------------
__global__ __launch_bounds__(256)
void flash_attn_kernel(const __bf16* __restrict__ qkv,
                       const float*  __restrict__ mask,
                       float*        __restrict__ out)
{
    __shared__ __bf16 Ks[32][72];        // [key][d]   -> B-frag for Q*K^T
    __shared__ __bf16 Vt[64][40];        // [d][key]   -> B-frag for P*V
    __shared__ __bf16 Ps[8][16][40];     // per-wave P re-layout (D -> A)

    const int tid  = threadIdx.x;
    const int wave = tid >> 5;
    const int lane = tid & 31;
    const int half = lane >> 4;
    const int r    = lane & 15;
    const int b    = blockIdx.x >> 4;
    const int h    = blockIdx.x & 15;
    const int qbase = blockIdx.y * 128 + wave * 16;

    const size_t rs = 3 * TDIM;
    const __bf16* qrow = qkv + (size_t)(b * TSEQ + qbase + r) * rs + h * THD;

    // Q fragments (A-layout), head-dim split 0..31 / 32..63; aligned b128 loads
    const __bf16* qp = qrow + half * 8;
    v8bf q00 = *(const v8bf*)(qp);
    v8bf q01 = *(const v8bf*)(qp + 16);
    v8bf q10 = *(const v8bf*)(qp + 32);
    v8bf q11 = *(const v8bf*)(qp + 48);
    v16bf a0 = __builtin_shufflevector(q00, q01, 0,1,2,3,4,5,6,7,8,9,10,11,12,13,14,15);
    v16bf a1 = __builtin_shufflevector(q10, q11, 0,1,2,3,4,5,6,7,8,9,10,11,12,13,14,15);

    v8f o[4] = {};
    float mrun[8], lrun[8];
#pragma unroll
    for (int i = 0; i < 8; ++i) { mrun[i] = -1e30f; lrun[i] = 0.0f; }

    // staging coordinates: 8 contiguous d-elements per thread
    const int kr = tid >> 3;         // key row 0..31
    const int c0 = (tid & 7) * 8;    // d chunk 0..56

    // LDS byte offset of this thread's K-tile destination (flat LDS addresses
    // carry the LDS offset in addr[31:0])
    const unsigned ks_lds = (unsigned)(size_t)&Ks[kr][c0];

    for (int j0 = 0; j0 < TSEQ; j0 += 32) {
        {
            const __bf16* src = qkv + (size_t)(b * TSEQ + j0 + kr) * rs + h * THD + c0;

            // K tile [32][64]: async DMA copy global -> LDS (no VGPR round-trip)
            asm volatile("global_load_async_to_lds_b128 %0, %1, off"
                         :: "v"(ks_lds), "v"(src + TDIM)
                         : "memory");

            // V tile transposed [64][32] (transpose can't be a straight copy)
            v8bf vv = *(const v8bf*)(src + 2 * TDIM);
#pragma unroll
            for (int j = 0; j < 8; ++j) Vt[c0 + j][kr] = vv[j];

            // make this wave's async K copies visible before the barrier
            asm volatile("s_wait_asynccnt 0x0" ::: "memory");
        }
        __syncthreads();

        // S = Q * K^T : two 16-key tiles x two K(d)-steps
        v8f s[2] = {};
#pragma unroll
        for (int t = 0; t < 2; ++t) {
            v16bf bk0 = load_frag(&Ks[t * 16][0], 72, 0,  lane);
            v16bf bk1 = load_frag(&Ks[t * 16][0], 72, 32, lane);
            s[t] = __builtin_amdgcn_wmma_f32_16x16x32_bf16(false, a0, false, bk0,
                                                           (short)0, s[t], false, false);
            s[t] = __builtin_amdgcn_wmma_f32_16x16x32_bf16(false, a1, false, bk1,
                                                           (short)0, s[t], false, false);
        }

        // scale (1/sqrt(64)) + additive mask
#pragma unroll
        for (int t = 0; t < 2; ++t)
#pragma unroll
            for (int i = 0; i < 8; ++i)
                s[t][i] = s[t][i] * 0.125f +
                          mask[(size_t)(qbase + i + 8 * half) * TSEQ + j0 + t * 16 + r];

        // online softmax update (row reductions across 16-lane halves)
#pragma unroll
        for (int i = 0; i < 8; ++i) {
            float v = fmaxf(s[0][i], s[1][i]);
#pragma unroll
            for (int off = 8; off >= 1; off >>= 1)
                v = fmaxf(v, __shfl_xor(v, off, 32));
            float mnew  = fmaxf(mrun[i], v);
            float alpha = __expf(mrun[i] - mnew);
            float p0 = __expf(s[0][i] - mnew);
            float p1 = __expf(s[1][i] - mnew);
            s[0][i] = p0; s[1][i] = p1;
            float sum = p0 + p1;
#pragma unroll
            for (int off = 8; off >= 1; off >>= 1)
                sum += __shfl_xor(sum, off, 32);
            lrun[i] = lrun[i] * alpha + sum;
            mrun[i] = mnew;
#pragma unroll
            for (int dt = 0; dt < 4; ++dt) o[dt][i] *= alpha;
        }

        // re-layout P (D layout -> A layout) through per-wave LDS
#pragma unroll
        for (int t = 0; t < 2; ++t)
#pragma unroll
            for (int i = 0; i < 8; ++i)
                Ps[wave][i + 8 * half][t * 16 + r] = (__bf16)s[t][i];

        v16bf ap = load_frag(&Ps[wave][0][0], 40, 0, lane);

        // O += P * V : 4 d-tiles of 16
#pragma unroll
        for (int dt = 0; dt < 4; ++dt) {
            v16bf bv = load_frag(&Vt[dt * 16][0], 40, 0, lane);
            o[dt] = __builtin_amdgcn_wmma_f32_16x16x32_bf16(false, ap, false, bv,
                                                            (short)0, o[dt], false, false);
        }
        __syncthreads();
    }

    // normalize (v_rcp_f32) and scatter to [b, q, h*64 + d] (f32)
    float inv[8];
#pragma unroll
    for (int i = 0; i < 8; ++i) inv[i] = __builtin_amdgcn_rcpf(lrun[i]);
#pragma unroll
    for (int dt = 0; dt < 4; ++dt)
#pragma unroll
        for (int i = 0; i < 8; ++i)
            out[(size_t)(b * TSEQ + qbase + i + 8 * half) * TDIM + h * THD
                + dt * 16 + r] = o[dt][i] * inv[i];
}

// ---------------------------------------------------------------------------
extern "C" void kernel_launch(void* const* d_in, const int* in_sizes, int n_in,
                              void* d_out, int out_size, void* d_ws, size_t ws_size,
                              hipStream_t stream)
{
    const float* seq  = (const float*)d_in[0];
    const float* mask = (const float*)d_in[1];
    const float* Wqkv = (const float*)d_in[2];
    const float* W1   = (const float*)d_in[3];
    const float* b1   = (const float*)d_in[4];
    const float* W2   = (const float*)d_in[5];
    const float* b2   = (const float*)d_in[6];

    // workspace layout (72 MiB total):
    //   [0, 24M)   qkv      bf16 [4096, 3072]
    //   [24M,40M)  attn_out f32  [4096, 1024]
    //   [40M,72M)  hid      f32  [4096, 2048]
    char*   ws       = (char*)d_ws;
    __bf16* qkv_bf   = (__bf16*)ws;
    float*  attn_out = (float*)(ws + (size_t)TROWS * 3 * TDIM * 2);
    float*  hid      = (float*)(ws + (size_t)TROWS * 3 * TDIM * 2
                                   + (size_t)TROWS * TDIM * 4);

    // 1) QKV projection: [4096,1024] @ [1024,3072] -> bf16
    gemm_wmma_kernel<0, 1, 0><<<dim3(TROWS / 64, (3 * TDIM) / 64), 256, 0, stream>>>(
        seq, Wqkv, nullptr, qkv_bf, TROWS, 3 * TDIM, TDIM);

    // 2) fused multi-head attention (online softmax)
    flash_attn_kernel<<<dim3(TBS * THEADS, TSEQ / 128), 256, 0, stream>>>(
        qkv_bf, mask, attn_out);

    // 3) FFN up-projection + SiLU: [4096,1024] @ [1024,2048]
    gemm_wmma_kernel<1, 0, 1><<<dim3(TROWS / 64, (2 * TDIM) / 64), 256, 0, stream>>>(
        attn_out, W1, b1, hid, TROWS, 2 * TDIM, TDIM);

    // 4) FFN down-projection + bias: [4096,2048] @ [2048,1024] -> d_out
    gemm_wmma_kernel<0, 0, 1><<<dim3(TROWS / 64, TDIM / 64), 256, 0, stream>>>(
        hid, W2, b2, d_out, TROWS, TDIM, 2 * TDIM);
}
